// GlobalGraphNet_40157944217634
// MI455X (gfx1250) — matrix operand order repaired and checked by reading, not compiled, not run on hardware
//
#include <hip/hip_runtime.h>

typedef _Float16 h8   __attribute__((ext_vector_type(8)));
typedef _Float16 v16h __attribute__((ext_vector_type(16)));
typedef float    v8f  __attribute__((ext_vector_type(8)));

#define TPB 256
#define EPSV 1e-5f

__device__ __forceinline__ float leaky01(float v){ return v > 0.0f ? v : 0.01f*v; }
__device__ __forceinline__ float leaky02(float v){ return v > 0.0f ? v : 0.2f*v; }
// monotonic float->uint key for atomicMax-based segment max
__device__ __forceinline__ unsigned fkey(float f){
  unsigned b = __float_as_uint(f);
  return (b & 0x80000000u) ? ~b : (b | 0x80000000u);
}
__device__ __forceinline__ float funkey(unsigned k){
  unsigned b = (k & 0x80000000u) ? (k & 0x7fffffffu) : ~k;
  return __uint_as_float(b);
}
#define SHUF16(lo, hi) __builtin_shufflevector(lo, hi, 0,1,2,3,4,5,6,7,8,9,10,11,12,13,14,15)

// ---------------- WMMA GEMM: C[rowsPad,64] = A[rowsPad,K](f16) @ B[K,64] ----
// BT is the f16 transposed weight: BT[n*K + k] = B[k*64 + n].
// One wave -> 16 rows x 64 cols, 4 accumulators, K stepped by 32.
// Software-pipelined: fragments for step k+32 are loaded while WMMAs for
// step k execute, so s_wait_loadcnt gates on loads issued a full iteration
// earlier instead of serializing memory->math.
template<int K>
__global__ void k_gemm16(const _Float16* __restrict__ A,
                         const _Float16* __restrict__ BT,
                         float* __restrict__ C, int rowsPad)
{
  const int lane  = threadIdx.x & 31;
  const int wave  = (int)((blockIdx.x * blockDim.x + threadIdx.x) >> 5);
  const int mbase = wave << 4;
  if (mbase >= rowsPad) return;            // wave-uniform: EXEC all-ones for WMMA
  const int g  = lane >> 4;                // lane group 0/1
  const int mr = lane & 15;

  const _Float16* arow = A  + (size_t)(mbase + mr) * K;
  const _Float16* br0  = BT + (size_t)(mr +  0) * K;
  const _Float16* br1  = BT + (size_t)(mr + 16) * K;
  const _Float16* br2  = BT + (size_t)(mr + 32) * K;
  const _Float16* br3  = BT + (size_t)(mr + 48) * K;

  // prologue: fragments for k0 = 0
  h8 alo = *(const h8*)(arow + g*16);
  h8 ahi = *(const h8*)(arow + g*16 + 8);
  int kb0 = g*16;
  h8 b0l = *(const h8*)(br0 + kb0), b0h = *(const h8*)(br0 + kb0 + 8);
  h8 b1l = *(const h8*)(br1 + kb0), b1h = *(const h8*)(br1 + kb0 + 8);
  h8 b2l = *(const h8*)(br2 + kb0), b2h = *(const h8*)(br2 + kb0 + 8);
  h8 b3l = *(const h8*)(br3 + kb0), b3h = *(const h8*)(br3 + kb0 + 8);

  v8f c0 = {}, c1 = {}, c2 = {}, c3 = {};
  for (int k0 = 0; k0 < K; k0 += 32) {
    // materialize current fragments
    v16h a  = SHUF16(alo, ahi);
    v16h b0 = SHUF16(b0l, b0h);
    v16h b1 = SHUF16(b1l, b1h);
    v16h b2 = SHUF16(b2l, b2h);
    v16h b3 = SHUF16(b3l, b3h);
    // prefetch next iteration's fragments (independent of the WMMAs below)
    const int kn = k0 + 32;
    if (kn < K) {
      alo = *(const h8*)(arow + kn + g*16);
      ahi = *(const h8*)(arow + kn + g*16 + 8);
      const int kb = kn + g*16;
      b0l = *(const h8*)(br0 + kb); b0h = *(const h8*)(br0 + kb + 8);
      b1l = *(const h8*)(br1 + kb); b1h = *(const h8*)(br1 + kb + 8);
      b2l = *(const h8*)(br2 + kb); b2h = *(const h8*)(br2 + kb + 8);
      b3l = *(const h8*)(br3 + kb); b3h = *(const h8*)(br3 + kb + 8);
    }
    c0 = __builtin_amdgcn_wmma_f32_16x16x32_f16(false, a, false, b0, (short)0, c0, false, false);
    c1 = __builtin_amdgcn_wmma_f32_16x16x32_f16(false, a, false, b1, (short)0, c1, false, false);
    c2 = __builtin_amdgcn_wmma_f32_16x16x32_f16(false, a, false, b2, (short)0, c2, false, false);
    c3 = __builtin_amdgcn_wmma_f32_16x16x32_f16(false, a, false, b3, (short)0, c3, false, false);
  }
  // D layout: VGPR v -> row mbase + 8*g + v, col = mr (+ntile*16)
  const int rb = mbase + 8*g;
#pragma unroll
  for (int v = 0; v < 8; ++v) {
    float* row = C + (size_t)(rb + v) * 64 + mr;
    row[0] = c0[v]; row[16] = c1[v]; row[32] = c2[v]; row[48] = c3[v];
  }
}

// ---------------- weight prep: f32 -> transposed f16 -----------------------
__global__ void k_prep_w(const float* __restrict__ Win, const float* __restrict__ gcnW,
                         const float* __restrict__ gatW,
                         _Float16* __restrict__ WinT, _Float16* __restrict__ WT10)
{
  int t = blockIdx.x * blockDim.x + threadIdx.x;
  if (t < 64*800) {                       // WinT[n*800+k] = Win[k*64+n]
    int n = t / 800, k = t % 800;
    WinT[t] = (_Float16)Win[k*64 + n];
  }
  int u = t - 64*800;
  if (u >= 0 && u < 10*64*64) {           // 5 gcn + 5 gat, each 64x64
    int i = u >> 12; int r = u & 4095; int n = r >> 6; int k = r & 63;
    const float* W = (i < 5) ? (gcnW + (size_t)i*4096) : (gatW + (size_t)(i-5)*4096);
    WT10[u] = (_Float16)W[k*64 + n];
  }
}

// ---------------- embedding gather -> featH f16 [Npad,800] -----------------
__global__ void k_embed(const int* __restrict__ x, const float* __restrict__ poi,
                        const float* __restrict__ cat, const float* __restrict__ lat,
                        const float* __restrict__ lon, _Float16* __restrict__ featH,
                        int N, long long total)
{
  long long t = (long long)blockIdx.x * blockDim.x + threadIdx.x;
  if (t >= total) return;
  int c = (int)(t % 800); long long n = t / 800;
  float v = 0.0f;
  if (n < N) {
    const int* xr = x + n*5;
    if      (c < 300) v = poi[(size_t)xr[0]*300 + c];
    else if (c < 400) v = cat[(size_t)xr[1]*100 + (c-300)];
    else if (c < 600) v = lat[(size_t)xr[3]*200 + (c-400)];
    else              v = lon[(size_t)xr[4]*200 + (c-600)];
  }
  featH[t] = (_Float16)v;
}

// ---------------- fills -----------------------------------------------------
__global__ void k_fill_f32(float* p, long long n, float v){
  long long t = (long long)blockIdx.x*blockDim.x + threadIdx.x; if (t < n) p[t] = v;
}
__global__ void k_fill_u32(unsigned* p, long long n, unsigned v){
  long long t = (long long)blockIdx.x*blockDim.x + threadIdx.x; if (t < n) p[t] = v;
}
__global__ void k_fill_from(float* p, const float* __restrict__ v, long long n){
  long long t = (long long)blockIdx.x*blockDim.x + threadIdx.x; if (t < n) p[t] = v[0];
}
__global__ void k_fill_bias64(float* p, const float* __restrict__ b, long long nelem){
  long long t = (long long)blockIdx.x*blockDim.x + threadIdx.x;
  if (t < nelem) p[t] = b[t & 63];
}

// ---------------- degrees ---------------------------------------------------
__global__ void k_deg(const int* __restrict__ ei, float* __restrict__ deg, int E){
  int e = blockIdx.x*blockDim.x + threadIdx.x;
  if (e < E) atomicAdd(&deg[ei[(long long)E + e]], 1.0f);
}
__global__ void k_rsqrt(const float* __restrict__ deg, float* __restrict__ dinv, int N){
  int t = blockIdx.x*blockDim.x + threadIdx.x;
  if (t < N) dinv[t] = rsqrtf(deg[t]);
}

// ---------------- GCN edge aggregation (wave per edge, 2 ch/lane) ----------
__global__ void k_gcn_agg(const int* __restrict__ ei, const float* __restrict__ dinv,
                          const float* __restrict__ hw, float* __restrict__ out,
                          int E, int N)
{
  long long t = (long long)blockIdx.x*blockDim.x + threadIdx.x;
  int lane = threadIdx.x & 31;
  long long e = t >> 5;
  if (e >= (long long)E + N) return;
  int s, d;
  if (e < E) { s = ei[e]; d = ei[(long long)E + e]; } else { s = d = (int)(e - E); }
  float w = dinv[s] * dinv[d];
  const float2 v = *(const float2*)(hw + (size_t)s*64 + lane*2);
  float* o = out + (size_t)d*64 + lane*2;
  atomicAdd(o,     w * v.x);
  atomicAdd(o + 1, w * v.y);
}

// ---------------- activation for layer-0 output -----------------------------
__global__ void k_act_h0(const float* __restrict__ g, float* __restrict__ h,
                         _Float16* __restrict__ hH, long long n){
  long long t = (long long)blockIdx.x*blockDim.x + threadIdx.x;
  if (t >= n) return;
  float v = leaky01(g[t]);
  h[t] = v; hH[t] = (_Float16)v;
}

// ---------------- graph-norm reductions -------------------------------------
__global__ void k_colsum(const float* __restrict__ g, float* __restrict__ stats, int N){
  int tid = blockIdx.x*blockDim.x + threadIdx.x;
  int c = tid & 63;
  int stride = (gridDim.x * blockDim.x) >> 6;
  float s = 0.0f;
  for (int n = tid >> 6; n < N; n += stride) s += g[(size_t)n*64 + c];
  atomicAdd(&stats[c], s);
}
__global__ void k_colvar(const float* __restrict__ g, float* __restrict__ stats,
                         const float* __restrict__ ms, int N){
  int tid = blockIdx.x*blockDim.x + threadIdx.x;
  int c = tid & 63;
  int stride = (gridDim.x * blockDim.x) >> 6;
  float mu = stats[c] / (float)N;
  float m  = ms[c];
  float s = 0.0f;
  for (int n = tid >> 6; n < N; n += stride) { float o = g[(size_t)n*64 + c] - m*mu; s += o*o; }
  atomicAdd(&stats[64 + c], s);
}
__global__ void k_norm_apply(const float* __restrict__ g, const float* __restrict__ stats,
                             const float* __restrict__ wv, const float* __restrict__ bv,
                             const float* __restrict__ ms, float* __restrict__ h,
                             _Float16* __restrict__ hH, int N){
  long long t = (long long)blockIdx.x*blockDim.x + threadIdx.x;
  if (t >= (long long)N*64) return;
  int c = (int)(t & 63);
  float mu  = stats[c] / (float)N;
  float var = stats[64 + c] / (float)N;
  float o = g[t] - ms[c]*mu;
  float y = leaky01(wv[c]*o*rsqrtf(var + EPSV) + bv[c]);
  float nh = h[t] + y;
  h[t] = nh; hH[t] = (_Float16)nh;
}

// ---------------- GAT pieces ------------------------------------------------
__global__ void k_dots(const float* __restrict__ hp, const float* __restrict__ asrc,
                       const float* __restrict__ adst, float* __restrict__ ssrc,
                       float* __restrict__ sdst, int N){
  int lane = threadIdx.x & 31;
  int n = (int)((blockIdx.x*blockDim.x + threadIdx.x) >> 5);
  if (n >= N) return;
  const float* r = hp + (size_t)n*64;
  float v0 = r[lane], v1 = r[32 + lane];
  float a = v0*asrc[lane] + v1*asrc[32+lane];
  float b = v0*adst[lane] + v1*adst[32+lane];
  for (int off = 16; off; off >>= 1) { a += __shfl_xor(a, off, 32); b += __shfl_xor(b, off, 32); }
  if (lane == 0) { ssrc[n] = a; sdst[n] = b; }
}
__global__ void k_gat_e1(const int* __restrict__ ei, const float* __restrict__ ssrc,
                         const float* __restrict__ sdst, float* __restrict__ al,
                         unsigned* __restrict__ mkey, int E, int N){
  long long e = (long long)blockIdx.x*blockDim.x + threadIdx.x;
  if (e >= (long long)E + N) return;
  int s, d;
  if (e < E) { s = ei[e]; d = ei[(long long)E + e]; } else { s = d = (int)(e - E); }
  float v = leaky02(ssrc[s] + sdst[d]);
  al[e] = v;
  atomicMax(&mkey[d], fkey(v));
}
__global__ void k_gat_e2(const int* __restrict__ ei, float* __restrict__ al,
                         const unsigned* __restrict__ mkey, float* __restrict__ ssum,
                         int E, int N){
  long long e = (long long)blockIdx.x*blockDim.x + threadIdx.x;
  if (e >= (long long)E + N) return;
  int d = (e < E) ? ei[(long long)E + e] : (int)(e - E);
  float p = __expf(al[e] - funkey(mkey[d]));
  al[e] = p;
  atomicAdd(&ssum[d], p);
}
__global__ void k_gat_agg(const int* __restrict__ ei, const float* __restrict__ p,
                          const float* __restrict__ ssum, const float* __restrict__ hp,
                          float* __restrict__ out, int E, int N){
  long long t = (long long)blockIdx.x*blockDim.x + threadIdx.x;
  int lane = threadIdx.x & 31;
  long long e = t >> 5;
  if (e >= (long long)E + N) return;
  int s, d;
  if (e < E) { s = ei[e]; d = ei[(long long)E + e]; } else { s = d = (int)(e - E); }
  float a = p[e] / ssum[d];
  const float2 v = *(const float2*)(hp + (size_t)s*64 + lane*2);
  float* o = out + (size_t)d*64 + lane*2;
  atomicAdd(o,     a * v.x);
  atomicAdd(o + 1, a * v.y);
}

// ---------------- final scalar GCN + FC head --------------------------------
__global__ void k_wout(const float* __restrict__ h, const float* __restrict__ Wout,
                       float* __restrict__ wout, int N){
  int lane = threadIdx.x & 31;
  int n = (int)((blockIdx.x*blockDim.x + threadIdx.x) >> 5);
  if (n >= N) return;
  const float* r = h + (size_t)n*64;
  float a = r[lane]*Wout[lane] + r[32+lane]*Wout[32+lane];
  for (int off = 16; off; off >>= 1) a += __shfl_xor(a, off, 32);
  if (lane == 0) wout[n] = a;
}
__global__ void k_agg_s(const int* __restrict__ ei, const float* __restrict__ dinv,
                        const float* __restrict__ wout, float* __restrict__ acc,
                        int E, int N){
  long long e = (long long)blockIdx.x*blockDim.x + threadIdx.x;
  if (e >= (long long)E + N) return;
  int s, d;
  if (e < E) { s = ei[e]; d = ei[(long long)E + e]; } else { s = d = (int)(e - E); }
  atomicAdd(&acc[d], dinv[s]*dinv[d]*wout[s]);
}
__global__ void k_leaky_vec(float* __restrict__ dst, const float* __restrict__ src, int n){
  int t = blockIdx.x*blockDim.x + threadIdx.x;
  if (t < n) dst[t] = leaky01(src[t]);
}
__global__ void k_fc1(const float* __restrict__ hs, const float* __restrict__ W1,
                      float* __restrict__ y1, int N){
  int j = threadIdx.x;                       // 128 threads
  float s = 0.0f;
  for (int r = blockIdx.x; r < N; r += gridDim.x) s += hs[r] * W1[(size_t)r*128 + j];
  atomicAdd(&y1[j], s);
}
__global__ void k_fc2(const float* __restrict__ y1, const float* __restrict__ b1,
                      const float* __restrict__ W2, const float* __restrict__ b2,
                      float* __restrict__ out){
  __shared__ float z[128];
  int k = threadIdx.x;
  z[k] = leaky01(y1[k] + b1[k]);
  __syncthreads();
  float s = b2[k];
  for (int j = 0; j < 128; ++j) s += z[j] * W2[j*128 + k];
  out[k] = s;
}

// ============================================================================
extern "C" void kernel_launch(void* const* d_in, const int* in_sizes, int n_in,
                              void* d_out, int out_size, void* d_ws, size_t ws_size,
                              hipStream_t stream)
{
  const int*   x       = (const int*)  d_in[0];
  const int*   ei      = (const int*)  d_in[1];
  const float* cat_emb = (const float*)d_in[3];
  const float* poi_emb = (const float*)d_in[4];
  const float* lat_emb = (const float*)d_in[5];
  const float* lon_emb = (const float*)d_in[6];
  const float* W_in    = (const float*)d_in[7];
  const float* b_in    = (const float*)d_in[8];
  const float* gcn_W   = (const float*)d_in[9];
  const float* gcn_b   = (const float*)d_in[10];
  const float* norm_w  = (const float*)d_in[11];
  const float* norm_b  = (const float*)d_in[12];
  const float* norm_ms = (const float*)d_in[13];
  const float* gat_W   = (const float*)d_in[14];
  const float* gat_as  = (const float*)d_in[15];
  const float* gat_ad  = (const float*)d_in[16];
  const float* gat_b   = (const float*)d_in[17];
  const float* W_out   = (const float*)d_in[18];
  const float* b_out   = (const float*)d_in[19];
  const float* fc_W1   = (const float*)d_in[20];
  const float* fc_b1   = (const float*)d_in[21];
  const float* fc_W2   = (const float*)d_in[22];
  const float* fc_b2   = (const float*)d_in[23];

  const int N = in_sizes[0] / 5;
  const int E = in_sizes[1] / 2;
  const int Npad = ((N + 127) / 128) * 128;     // rows multiple of 128 (8 waves/blk)
  const long long EN = (long long)E + N;

  // -------- workspace carve-out (256B aligned slabs) --------
  char* w = (char*)d_ws;
  auto carve = [&](size_t bytes) -> void* {
    void* p = (void*)w; w += (bytes + 255) & ~(size_t)255; return p;
  };
  _Float16* featH = (_Float16*)carve((size_t)Npad * 800 * 2);
  _Float16* hH    = (_Float16*)carve((size_t)Npad * 64 * 2);
  float*    h     = (float*)   carve((size_t)Npad * 64 * 4);
  float*    g     = (float*)   carve((size_t)Npad * 64 * 4);
  float*    hw    = (float*)   carve((size_t)Npad * 64 * 4);
  _Float16* WinT  = (_Float16*)carve((size_t)64 * 800 * 2);
  _Float16* WT10  = (_Float16*)carve((size_t)10 * 64 * 64 * 2);
  float*    deg   = (float*)   carve((size_t)N * 4);
  float*    dinv  = (float*)   carve((size_t)N * 4);
  float*    stats = (float*)   carve((size_t)128 * 4);
  float*    al    = (float*)   carve((size_t)EN * 4);
  unsigned* mkey  = (unsigned*)carve((size_t)N * 4);
  float*    ssum  = (float*)   carve((size_t)N * 4);
  float*    ssrc  = (float*)   carve((size_t)N * 4);
  float*    sdst  = (float*)   carve((size_t)N * 4);
  float*    wout  = (float*)   carve((size_t)N * 4);
  float*    hsacc = (float*)   carve((size_t)N * 4);
  float*    hs    = (float*)   carve((size_t)N * 4);
  float*    y1    = (float*)   carve((size_t)128 * 4);
  (void)ws_size; (void)n_in; (void)out_size;

  auto nb = [](long long n) { return (unsigned)((n + TPB - 1) / TPB); };
  const unsigned gemmBlocks = (unsigned)(Npad / 128);

  // -------- prep: weights, embeddings, degrees --------
  k_prep_w<<<nb(64*800 + 10*64*64), TPB, 0, stream>>>(W_in, gcn_W, gat_W, WinT, WT10);
  k_embed<<<nb((long long)Npad * 800), TPB, 0, stream>>>(x, poi_emb, cat_emb, lat_emb,
                                                         lon_emb, featH, N, (long long)Npad*800);
  k_fill_f32<<<nb(N), TPB, 0, stream>>>(deg, N, 1.0f);          // self-loop
  k_deg<<<nb(E), TPB, 0, stream>>>(ei, deg, E);
  k_rsqrt<<<nb(N), TPB, 0, stream>>>(deg, dinv, N);

  // -------- input GCN: h0 = leaky(agg(feat @ W_in) + b_in) --------
  k_gemm16<800><<<gemmBlocks, TPB, 0, stream>>>(featH, WinT, hw, Npad);
  k_fill_bias64<<<nb((long long)Npad*64), TPB, 0, stream>>>(g, b_in, (long long)Npad*64);
  k_gcn_agg<<<nb(EN*32), TPB, 0, stream>>>(ei, dinv, hw, g, E, N);
  k_act_h0<<<nb((long long)Npad*64), TPB, 0, stream>>>(g, h, hH, (long long)Npad*64);

  for (int i = 0; i < 5; ++i) {
    const float* nw = norm_w + i*64; const float* nbv = norm_b + i*64;
    const float* nm = norm_ms + i*64;

    // ---- GCN branch ----
    k_gemm16<64><<<gemmBlocks, TPB, 0, stream>>>(hH, WT10 + (size_t)i*4096, hw, Npad);
    k_fill_bias64<<<nb((long long)Npad*64), TPB, 0, stream>>>(g, gcn_b + i*64, (long long)Npad*64);
    k_gcn_agg<<<nb(EN*32), TPB, 0, stream>>>(ei, dinv, hw, g, E, N);
    k_fill_f32<<<1, 128, 0, stream>>>(stats, 128, 0.0f);
    k_colsum<<<240, TPB, 0, stream>>>(g, stats, N);
    k_colvar<<<240, TPB, 0, stream>>>(g, stats, nm, N);
    k_norm_apply<<<nb((long long)N*64), TPB, 0, stream>>>(g, stats, nw, nbv, nm, h, hH, N);

    // ---- GAT branch ----
    k_gemm16<64><<<gemmBlocks, TPB, 0, stream>>>(hH, WT10 + (size_t)(5+i)*4096, hw, Npad);
    k_dots<<<nb((long long)N*32), TPB, 0, stream>>>(hw, gat_as + i*64, gat_ad + i*64, ssrc, sdst, N);
    k_fill_u32<<<nb(N), TPB, 0, stream>>>(mkey, N, 0u);
    k_gat_e1<<<nb(EN), TPB, 0, stream>>>(ei, ssrc, sdst, al, mkey, E, N);
    k_fill_f32<<<nb(N), TPB, 0, stream>>>(ssum, N, 0.0f);
    k_gat_e2<<<nb(EN), TPB, 0, stream>>>(ei, al, mkey, ssum, E, N);
    k_fill_bias64<<<nb((long long)Npad*64), TPB, 0, stream>>>(g, gat_b + i*64, (long long)Npad*64);
    k_gat_agg<<<nb(EN*32), TPB, 0, stream>>>(ei, al, ssum, hw, g, E, N);
    k_fill_f32<<<1, 128, 0, stream>>>(stats, 128, 0.0f);
    k_colsum<<<240, TPB, 0, stream>>>(g, stats, N);
    k_colvar<<<240, TPB, 0, stream>>>(g, stats, nm, N);
    k_norm_apply<<<nb((long long)N*64), TPB, 0, stream>>>(g, stats, nw, nbv, nm, h, hH, N);
  }

  // -------- final scalar GCN + FC head --------
  k_wout<<<nb((long long)N*32), TPB, 0, stream>>>(h, W_out, wout, N);
  k_fill_from<<<nb(N), TPB, 0, stream>>>(hsacc, b_out, N);
  k_agg_s<<<nb(EN), TPB, 0, stream>>>(ei, dinv, wout, hsacc, E, N);
  k_leaky_vec<<<nb(N), TPB, 0, stream>>>(hs, hsacc, N);
  k_fill_f32<<<1, 128, 0, stream>>>(y1, 128, 0.0f);
  k_fc1<<<512, 128, 0, stream>>>(hs, fc_W1, y1, N);
  k_fc2<<<1, 128, 0, stream>>>(y1, fc_b1, fc_W2, fc_b2, (float*)d_out);
}
